// ContrastiveLoss_40372692582790
// MI455X (gfx1250) — matrix operand order repaired and checked by reading, not compiled, not run on hardware
//
#include <hip/hip_runtime.h>
#include <math.h>

// Problem constants (fixed by the reference).
#define BATCH 4096
#define DIM   512
#define NROW  8192     // 2*BATCH
#define NNEG  4
#define NTILE (NROW / 16)        // 512 tiles of 16 query rows
#define WAVES_PER_BLOCK 8

typedef __attribute__((ext_vector_type(16))) _Float16 v16h;
typedef __attribute__((ext_vector_type(8)))  float    v8f;

union HV16 { _Float16 h[16]; v16h v; };

static __device__ __forceinline__ const float* row_ptr(const float* zi, const float* zj, int n) {
  return (n < BATCH) ? (zi + (size_t)n * DIM) : (zj + (size_t)(n - BATCH) * DIM);
}

// Load 8 consecutive f32 (32B, aligned) and convert to f16.
static __device__ __forceinline__ void cvt8(const float* __restrict__ p, _Float16* d) {
  const float4* p4 = reinterpret_cast<const float4*>(p);
  float4 a = p4[0];
  float4 b = p4[1];
  d[0] = (_Float16)a.x; d[1] = (_Float16)a.y; d[2] = (_Float16)a.z; d[3] = (_Float16)a.w;
  d[4] = (_Float16)b.x; d[5] = (_Float16)b.y; d[6] = (_Float16)b.z; d[7] = (_Float16)b.w;
}

// ---------------------------------------------------------------------------
// Kernel 1: inverse row norms. One wave32 per row; lane reads 16 floats.
// ---------------------------------------------------------------------------
__global__ void __launch_bounds__(256) norm_kernel(const float* __restrict__ zi,
                                                   const float* __restrict__ zj,
                                                   float* __restrict__ rinv) {
  int gid  = blockIdx.x * 256 + threadIdx.x;
  int row  = gid >> 5;
  int lane = gid & 31;
  const float4* p = reinterpret_cast<const float4*>(row_ptr(zi, zj, row));
  float s = 0.f;
#pragma unroll
  for (int w = 0; w < 4; ++w) {
    float4 v = p[w * 32 + lane];           // coalesced 512B per step per wave
    s = fmaf(v.x, v.x, s); s = fmaf(v.y, v.y, s);
    s = fmaf(v.z, v.z, s); s = fmaf(v.w, v.w, s);
  }
#pragma unroll
  for (int off = 16; off > 0; off >>= 1) s += __shfl_xor(s, off, 32);
  if (lane == 0) rinv[row] = 1.0f / fmaxf(sqrtf(s), 1e-8f);
}

// ---------------------------------------------------------------------------
// Kernel 2: per-row logits via WMMA + log-softmax NLL.
// One wave per tile of 16 query rows. For each key-set j (0 = positive,
// 1..4 = sampled negatives) accumulate D = A(keys 16x32) * B(queries 32x16)
// over K=512; the diagonal of D holds the needed dot products.
// ---------------------------------------------------------------------------
__global__ void __launch_bounds__(256) logits_kernel(const float* __restrict__ zi,
                                                     const float* __restrict__ zj,
                                                     const int*   __restrict__ neg_idx,
                                                     const float* __restrict__ rinv,
                                                     float*       __restrict__ tile_sums) {
  __shared__ float diag_lds[WAVES_PER_BLOCK][5][256];
  const float INV_T = (float)(1.0 / (0.5 + 1e-8));   // 1/(TEMPERATURE+EPSILON)

  const int wave = threadIdx.x >> 5;
  const int lane = threadIdx.x & 31;
  const int n15  = lane & 15;      // A-row / B-col / C-col owned by this lane
  const int half = lane >> 4;      // K-half selector within fragments
  const int tile = blockIdx.x * WAVES_PER_BLOCK + wave;
  const int qi   = tile * 16 + n15;              // query row for this lane

  // Gathered key rows per the reference's index math (uniform per lane pair).
  int key[5];
  key[0] = (qi + BATCH) & (NROW - 1);            // positive: (i+B) % N
#pragma unroll
  for (int r = 0; r < NNEG; ++r) {
    int raw = neg_idx[qi * NNEG + r];            // in [0, N-1)
    key[r + 1] = raw + ((raw >= qi) ? 1 : 0);    // skip diagonal column
  }
  const float* qp = row_ptr(zi, zj, qi);
  const float* kp[5];
#pragma unroll
  for (int j = 0; j < 5; ++j) kp[j] = row_ptr(zi, zj, key[j]);

  v8f c[5];
  const v8f vzero = {0.f, 0.f, 0.f, 0.f, 0.f, 0.f, 0.f, 0.f};
#pragma unroll
  for (int j = 0; j < 5; ++j) c[j] = vzero;

  // K loop: 16 steps of K=32. Fragment layouts per ISA 7.12.2:
  //  A 16x32 f16 : lane(M=n15, half h) holds K = {k0+8h .. k0+8h+7} and {k0+16+8h ..}
  //  B 32x16 f16 : lane(N=n15, half h) holds K = {k0+8h .. k0+8h+7} and {k0+16+8h ..}
  for (int k0 = 0; k0 < DIM; k0 += 32) {
    HV16 bf;
    cvt8(qp + k0 + half * 8,      bf.h);
    cvt8(qp + k0 + 16 + half * 8, bf.h + 8);
#pragma unroll
    for (int j = 0; j < 5; ++j) {
      HV16 af;
      cvt8(kp[j] + k0 + half * 8,      af.h);
      cvt8(kp[j] + k0 + 16 + half * 8, af.h + 8);
      // D = A*B + C  (f32 accumulate)
      c[j] = __builtin_amdgcn_wmma_f32_16x16x32_f16(
          /*neg_a=*/false, af.v, /*neg_b=*/false, bf.v,
          /*c_mod=*/(short)0, c[j], /*reuse_a=*/false, /*reuse_b=*/false);
    }
  }

  // Stage C tiles to LDS to pull out the diagonal (C layout: lanes 0-15 hold
  // M=v, N=lane; lanes 16-31 hold M=v+8, N=lane-16).
#pragma unroll
  for (int j = 0; j < 5; ++j) {
#pragma unroll
    for (int v = 0; v < 8; ++v)
      diag_lds[wave][j][(v + half * 8) * 16 + n15] = c[j][v];
  }
  __syncthreads();

  const float ri = rinv[qi];
  float l[5];
#pragma unroll
  for (int j = 0; j < 5; ++j)
    l[j] = diag_lds[wave][j][n15 * 16 + n15] * ri * rinv[key[j]] * INV_T;

  // logits = [pos, neg0..neg3], label 0 -> nll = -log_softmax(l)[0]
  float m = l[0];
#pragma unroll
  for (int j = 1; j < 5; ++j) m = fmaxf(m, l[j]);
  float s = 0.f;
#pragma unroll
  for (int j = 0; j < 5; ++j) s += expf(l[j] - m);
  float nll = (half == 0) ? (m + logf(s) - l[0]) : 0.f;  // halves are duplicates

#pragma unroll
  for (int off = 16; off > 0; off >>= 1) nll += __shfl_xor(nll, off, 32);
  if (lane == 0) tile_sums[tile] = nll;
}

// ---------------------------------------------------------------------------
// Kernel 3: deterministic fixed-order reduction of 512 tile sums -> mean loss.
// ---------------------------------------------------------------------------
__global__ void __launch_bounds__(256) reduce_kernel(const float* __restrict__ tile_sums,
                                                     float* __restrict__ out) {
  __shared__ float sd[256];
  int t = threadIdx.x;
  sd[t] = tile_sums[t] + tile_sums[t + 256];
  __syncthreads();
  for (int off = 128; off > 0; off >>= 1) {
    if (t < off) sd[t] += sd[t + off];
    __syncthreads();
  }
  if (t == 0) out[0] = sd[0] * (1.0f / (float)NROW);
}

extern "C" void kernel_launch(void* const* d_in, const int* in_sizes, int n_in,
                              void* d_out, int out_size, void* d_ws, size_t ws_size,
                              hipStream_t stream) {
  (void)in_sizes; (void)n_in; (void)out_size; (void)ws_size;
  const float* zi  = (const float*)d_in[0];
  const float* zj  = (const float*)d_in[1];
  const int*   neg = (const int*)d_in[2];
  float* out       = (float*)d_out;

  float* rinv      = (float*)d_ws;        // NROW floats
  float* tile_sums = rinv + NROW;         // NTILE floats

  norm_kernel  <<<NROW / 8,                 256, 0, stream>>>(zi, zj, rinv);
  logits_kernel<<<NTILE / WAVES_PER_BLOCK,  256, 0, stream>>>(zi, zj, neg, rinv, tile_sums);
  reduce_kernel<<<1,                        256, 0, stream>>>(tile_sums, out);
}
